// MultiHeadAttention_88261577932903
// MI455X (gfx1250) — matrix operand order repaired and checked by reading, not compiled, not run on hardware
//
#include <hip/hip_runtime.h>
#include <hip/hip_bf16.h>

#define N_HEADS   16
#define HEAD_SIZE 64
#define EMBED     1024
#define CTX       2048
#define BATCH     4

typedef __attribute__((ext_vector_type(4)))  unsigned int u32x4;
typedef __attribute__((ext_vector_type(16))) __bf16       bf16x16;
typedef __attribute__((ext_vector_type(8)))  float        f32x8;

union Frag {
    bf16x16 v;
    u32x4   q[2];
};

__device__ __forceinline__ unsigned short f2bf(float f) {
    unsigned u = __float_as_uint(f);
    unsigned r = u + 0x7FFFu + ((u >> 16) & 1u);   // round-to-nearest-even
    return (unsigned short)(r >> 16);
}

// ---------------------------------------------------------------------------
// WMMA fragment loaders. Layouts per CDNA5 ISA 7.12.2 (wave32):
//   A (16x32 bf16, MxK): lane l -> row m=l&15; halves K=kb..kb+7, 16+kb..16+kb+7,
//       kb = 8*(l>=16)  -> 2x b128
//   B (32x16 bf16, KxN), source stored as B^T row-major [N][K]:
//       lane l -> col n=l&15; halves K=kh..kh+7, 16+kh..16+kh+7, kh = 8*(l>=16)  -> 2x b128
// ---------------------------------------------------------------------------
__device__ __forceinline__ Frag load_frag_a(const unsigned short* base, int ld) {
    const int lane = threadIdx.x & 31;
    const int m  = lane & 15;
    const int kb = (lane >> 4) << 3;
    const unsigned short* p = base + (size_t)m * ld + kb;
    Frag f;
    f.q[0] = *(const u32x4*)(p);
    f.q[1] = *(const u32x4*)(p + 16);
    return f;
}

__device__ __forceinline__ Frag load_frag_b(const unsigned short* bT, int ld) {
    const int lane = threadIdx.x & 31;
    const int n  = lane & 15;
    const int kh = (lane >> 4) << 3;
    const unsigned short* p = bT + (size_t)n * ld + kh;
    Frag f;
    f.q[0] = *(const u32x4*)(p);
    f.q[1] = *(const u32x4*)(p + 16);
    return f;
}

__device__ __forceinline__ f32x8 wmma_bf16(const Frag& a, const Frag& b, f32x8 c) {
    return __builtin_amdgcn_wmma_f32_16x16x32_bf16(false, a.v, false, b.v,
                                                   (short)0, c, false, false);
}

// ---------------------------------------------------------------------------
// Async global->LDS staging of one 64x32 bf16 B^T panel (4 KB).
// 128 threads: each copies 32 B (two b128) via the ASYNCcnt-tracked path.
// ---------------------------------------------------------------------------
__device__ __forceinline__ void async_panel_64x32(const unsigned short* gbase,
                                                  int ld, unsigned lds_byte) {
    const int tid  = threadIdx.x;          // 0..127
    const int row  = tid >> 1;             // 0..63
    const int half = tid & 1;              // 0..1 (32B halves of a 64B row)
    unsigned voff = (unsigned)((row * ld + half * 16) * 2);   // bytes
    unsigned la   = lds_byte + (unsigned)(row * 64 + half * 32);
    asm volatile("global_load_async_to_lds_b128 %0, %1, %2"
                 :: "v"(la), "v"(voff), "s"(gbase) : "memory");
    asm volatile("global_load_async_to_lds_b128 %0, %1, %2 offset:16"
                 :: "v"(la + 16), "v"(voff), "s"(gbase) : "memory");
}

// One K-step of the panel GEMM: 4 WMMAs from an LDS panel buffer.
__device__ __forceinline__ void gemm_step(const Frag& a, const unsigned short* pl,
                                          f32x8 acc[4]) {
#pragma unroll
    for (int j = 0; j < 4; j++) {
        Frag bf = load_frag_b(pl + j * 16 * 32, 32);   // ds_load_b128 x2
        acc[j] = wmma_bf16(a, bf, acc[j]);
    }
}

// ---------------------------------------------------------------------------
// Shared K-loop: acc[0..3] (16x64 tile) += A[16 x 1024] * B, with the B^T
// panel [64 x 1024] staged through double-buffered LDS by async copies.
// Manually 2x unrolled with named fragment buffers a0/a1 so no fragment
// copies (and no WMMA->VALU hazard NOPs) are needed.  Uniform control flow.
// ---------------------------------------------------------------------------
__device__ __forceinline__ void gemm16x64_panel(const unsigned short* aBase,
                                                const unsigned short* bPanel,
                                                unsigned short* lds,  // 2*64*32
                                                f32x8 acc[4]) {
    const unsigned lbase = (unsigned)(uintptr_t)lds;     // LDS byte address
    const unsigned bufB  = 64 * 32 * 2;                  // bytes per buffer

    async_panel_64x32(bPanel, EMBED, lbase);             // panel(e=0) -> buf0
    Frag a0 = load_frag_a(aBase, EMBED);
    Frag a1;
    asm volatile("s_wait_asynccnt 0" ::: "memory");
    __syncthreads();

    for (int e = 0; e < EMBED; e += 64) {
        // ---- phase 0: consume buf0/a0; prefetch e+32 -> buf1/a1
        if (e + 32 < EMBED) {
            async_panel_64x32(bPanel + e + 32, EMBED, lbase + bufB);
            a1 = load_frag_a(aBase + e + 32, EMBED);
        }
        gemm_step(a0, lds, acc);
        if (e + 32 < EMBED)
            asm volatile("s_wait_asynccnt 0" ::: "memory");
        __syncthreads();
        if (e + 32 >= EMBED) break;
        // ---- phase 1: consume buf1/a1; prefetch e+64 -> buf0/a0
        if (e + 64 < EMBED) {
            async_panel_64x32(bPanel + e + 64, EMBED, lbase);
            a0 = load_frag_a(aBase + e + 64, EMBED);
        }
        gemm_step(a1, lds + 64 * 32, acc);
        if (e + 64 < EMBED)
            asm volatile("s_wait_asynccnt 0" ::: "memory");
        __syncthreads();
    }
}

// ---------------------------------------------------------------------------
// fp32 -> bf16 elementwise convert
// ---------------------------------------------------------------------------
__global__ void cvt_f32_bf16(const float* __restrict__ src,
                             unsigned short* __restrict__ dst, int n) {
    int i = blockIdx.x * blockDim.x + threadIdx.x;
    if (i < n) dst[i] = f2bf(src[i]);
}

// Batched transpose+convert: src [R,C] fp32 -> dst [C,R] bf16 (per grid.z batch)
__global__ void transpose_cvt(const float* __restrict__ src,
                              unsigned short* __restrict__ dst, int R, int C) {
    size_t boff = (size_t)blockIdx.z * R * C;
    int i = blockIdx.x * blockDim.x + threadIdx.x;
    if (i < R * C) {
        int r = i / C, c = i % C;
        dst[boff + (size_t)c * R + r] = f2bf(src[boff + i]);
    }
}

// ---------------------------------------------------------------------------
// QKV projection: per wave, one 16(s) x 64(d) tile of q/k/v for one (b,h).
// All 4 waves share the same weight panel -> staged via async LDS.
// v is written transposed ([B,H,D,S]) for the P*V B-operand.
// grid: (S/64, H, B*3), block 128 (4 waves)
// ---------------------------------------------------------------------------
__global__ __launch_bounds__(128) void qkv_proj(
        const unsigned short* __restrict__ xbf,
        const unsigned short* __restrict__ wqT,
        const unsigned short* __restrict__ wkT,
        const unsigned short* __restrict__ wvT,
        unsigned short* __restrict__ qb,
        unsigned short* __restrict__ kb,
        unsigned short* __restrict__ vt) {
    __shared__ __align__(16) unsigned short panel[2][64 * 32];

    const int lane = threadIdx.x & 31, wave = threadIdx.x >> 5;
    const int srow  = (blockIdx.x * 4 + wave) * 16;
    const int h     = blockIdx.y;
    const int b     = blockIdx.z / 3, which = blockIdx.z % 3;
    const unsigned short* wT =
        (which == 0 ? wqT : which == 1 ? wkT : wvT) + (size_t)h * HEAD_SIZE * EMBED;
    const unsigned short* xa = xbf + (size_t)(b * CTX + srow) * EMBED;

    f32x8 acc[4] = {};
    gemm16x64_panel(xa, wT, &panel[0][0], acc);

    const int n = lane & 15, rbase = (lane >> 4) * 8;
    if (which < 2) {
        unsigned short* dst =
            (which == 0 ? qb : kb) + ((size_t)(b * N_HEADS + h) * CTX + srow) * HEAD_SIZE;
#pragma unroll
        for (int j = 0; j < 4; j++)
#pragma unroll
            for (int r = 0; r < 8; r++)
                dst[(size_t)(rbase + r) * HEAD_SIZE + j * 16 + n] = f2bf(acc[j][r]);
    } else {
        unsigned short* dst = vt + (size_t)(b * N_HEADS + h) * HEAD_SIZE * CTX;
#pragma unroll
        for (int j = 0; j < 4; j++)
#pragma unroll
            for (int r = 0; r < 8; r++)
                dst[(size_t)(j * 16 + n) * CTX + srow + rbase + r] = f2bf(acc[j][r]);
    }
}

// ---------------------------------------------------------------------------
// Causal flash attention: one wave per 16-query tile, 32 keys per step.
// No barriers (divergent causal trip counts) -> per-wave LDS P staging.
// grid: (S/64, H, B), block 128 (4 waves). Output -> cat [B*S, H*D] bf16.
// ---------------------------------------------------------------------------
__global__ __launch_bounds__(128) void flash_attn(
        const unsigned short* __restrict__ qg,
        const unsigned short* __restrict__ kg,
        const unsigned short* __restrict__ vt,
        unsigned short* __restrict__ cat) {
    __shared__ __align__(16) unsigned short p_lds[4][16 * 32];

    const int lane = threadIdx.x & 31, wave = threadIdx.x >> 5;
    const int qrow = (blockIdx.x * 4 + wave) * 16;
    const int h = blockIdx.y, b = blockIdx.z;
    const size_t bh = (size_t)(b * N_HEADS + h);
    const unsigned short* qp = qg + bh * CTX * HEAD_SIZE;
    const unsigned short* kp = kg + bh * CTX * HEAD_SIZE;
    const unsigned short* vp = vt + bh * HEAD_SIZE * CTX;

    const Frag aq0 = load_frag_a(qp + (size_t)qrow * HEAD_SIZE, HEAD_SIZE);
    const Frag aq1 = load_frag_a(qp + (size_t)qrow * HEAD_SIZE + 32, HEAD_SIZE);

    f32x8 oacc[4] = {};
    float mst[8], lst[8];
#pragma unroll
    for (int r = 0; r < 8; r++) { mst[r] = -__builtin_inff(); lst[r] = 0.f; }

    const int n = lane & 15, rbase = (lane >> 4) * 8;
    const int kend = qrow + 16;   // keys 0 .. qrow+15 can be unmasked

    for (int k0 = 0; k0 < kend; k0 += 32) {
        if (k0 + 32 < kend) {  // prefetch next K tile -> global_prefetch_b8
            __builtin_prefetch(kp + (size_t)(k0 + 32) * HEAD_SIZE, 0, 1);
        }
        // ---- scores: S[16q x 32k] as two 16-key C fragments (K-dim = D = 64)
        f32x8 s[2];
#pragma unroll
        for (int c = 0; c < 2; c++) {
            Frag b0 = load_frag_b(kp + (size_t)(k0 + c * 16) * HEAD_SIZE, HEAD_SIZE);
            Frag b1 = load_frag_b(kp + (size_t)(k0 + c * 16) * HEAD_SIZE + 32, HEAD_SIZE);
            f32x8 t = {};
            t = wmma_bf16(aq0, b0, t);
            t = wmma_bf16(aq1, b1, t);
            s[c] = t;
        }
        // ---- issue V fragment loads now; they complete under the softmax VALU
        Frag bv[4];
#pragma unroll
        for (int j = 0; j < 4; j++)
            bv[j] = load_frag_b(vp + (size_t)(j * 16) * CTX + k0, CTX);
        // ---- scale + causal mask (C layout: row = rbase+r, col = 16c+n)
#pragma unroll
        for (int c = 0; c < 2; c++)
#pragma unroll
            for (int r = 0; r < 8; r++) {
                int key = k0 + c * 16 + n;
                int qi  = qrow + rbase + r;
                float v = s[c][r] * 0.125f;   // HEAD_SIZE^-0.5
                s[c][r] = (key <= qi) ? v : -__builtin_inff();
            }
        // ---- online softmax (row reductions across the 16 lanes owning a row)
#pragma unroll
        for (int r = 0; r < 8; r++) {
            float t = fmaxf(s[0][r], s[1][r]);
            t = fmaxf(t, __shfl_xor(t, 1, 32));
            t = fmaxf(t, __shfl_xor(t, 2, 32));
            t = fmaxf(t, __shfl_xor(t, 4, 32));
            t = fmaxf(t, __shfl_xor(t, 8, 32));
            float mnew = fmaxf(mst[r], t);
            float corr = __expf(mst[r] - mnew);
            mst[r] = mnew;
            lst[r] *= corr;
#pragma unroll
            for (int j = 0; j < 4; j++) oacc[j][r] *= corr;
            float p0 = __expf(s[0][r] - mnew);
            float p1 = __expf(s[1][r] - mnew);
            s[0][r] = p0; s[1][r] = p1;
            float rs = p0 + p1;
            rs += __shfl_xor(rs, 1, 32);
            rs += __shfl_xor(rs, 2, 32);
            rs += __shfl_xor(rs, 4, 32);
            rs += __shfl_xor(rs, 8, 32);
            lst[r] += rs;
        }
        // ---- P (C layout) -> LDS bf16 [16][32] -> reload in A layout
        unsigned short* pl = p_lds[wave];
#pragma unroll
        for (int c = 0; c < 2; c++)
#pragma unroll
            for (int r = 0; r < 8; r++)
                pl[(rbase + r) * 32 + c * 16 + n] = f2bf(s[c][r]);
        asm volatile("s_wait_dscnt 0" ::: "memory");  // cross-lane LDS RAW
        Frag ap = load_frag_a(pl, 32);
        // ---- O += P * V  (B from V^T [D,S], fragments preloaded above)
#pragma unroll
        for (int j = 0; j < 4; j++)
            oacc[j] = wmma_bf16(ap, bv[j], oacc[j]);
    }
    // ---- epilogue: O /= l ; write concat [B*S, H*D]
    unsigned short* dst =
        cat + ((size_t)(b * CTX + qrow)) * (N_HEADS * HEAD_SIZE) + h * HEAD_SIZE;
#pragma unroll
    for (int j = 0; j < 4; j++)
#pragma unroll
        for (int r = 0; r < 8; r++) {
            float o = oacc[j][r] / lst[r];
            dst[(size_t)(rbase + r) * (N_HEADS * HEAD_SIZE) + j * 16 + n] = f2bf(o);
        }
}

// ---------------------------------------------------------------------------
// Output projection: out[8192,1024] = cat[8192,1024] @ Wo  (B from Wo^T)
// grid: (M/64, E/64), block 128 (4 waves), fp32 output.
// ---------------------------------------------------------------------------
__global__ __launch_bounds__(128) void out_proj(
        const unsigned short* __restrict__ cat,
        const unsigned short* __restrict__ woT,
        float* __restrict__ out) {
    __shared__ __align__(16) unsigned short panel[2][64 * 32];

    const int lane = threadIdx.x & 31, wave = threadIdx.x >> 5;
    const int m0 = (blockIdx.x * 4 + wave) * 16;
    const int n0 = blockIdx.y * 64;

    f32x8 acc[4] = {};
    gemm16x64_panel(cat + (size_t)m0 * EMBED,
                    woT + (size_t)n0 * EMBED, &panel[0][0], acc);

    const int n = lane & 15, rbase = (lane >> 4) * 8;
#pragma unroll
    for (int j = 0; j < 4; j++)
#pragma unroll
        for (int r = 0; r < 8; r++)
            out[(size_t)(m0 + rbase + r) * EMBED + n0 + j * 16 + n] = acc[j][r];
}

// ---------------------------------------------------------------------------
extern "C" void kernel_launch(void* const* d_in, const int* in_sizes, int n_in,
                              void* d_out, int out_size, void* d_ws, size_t ws_size,
                              hipStream_t stream) {
    const float* x  = (const float*)d_in[0];
    const float* Wq = (const float*)d_in[1];
    const float* Wk = (const float*)d_in[2];
    const float* Wv = (const float*)d_in[3];
    const float* Wo = (const float*)d_in[4];
    float* out = (float*)d_out;

    char* ws = (char*)d_ws;
    size_t off = 0;
    auto alloc = [&](size_t bytes) -> void* {
        void* p = ws + off;
        off += (bytes + 255) & ~(size_t)255;
        return p;
    };
    const size_t BS = (size_t)BATCH * CTX;                  // 8192
    unsigned short* xbf = (unsigned short*)alloc(BS * EMBED * 2);                   // 16MB
    unsigned short* wqT = (unsigned short*)alloc((size_t)N_HEADS * HEAD_SIZE * EMBED * 2);
    unsigned short* wkT = (unsigned short*)alloc((size_t)N_HEADS * HEAD_SIZE * EMBED * 2);
    unsigned short* wvT = (unsigned short*)alloc((size_t)N_HEADS * HEAD_SIZE * EMBED * 2);
    unsigned short* woT = (unsigned short*)alloc((size_t)EMBED * EMBED * 2);        // 2MB
    unsigned short* qb  = (unsigned short*)alloc(BS * N_HEADS * HEAD_SIZE * 2);     // [B,H,S,D]
    unsigned short* kb  = (unsigned short*)alloc(BS * N_HEADS * HEAD_SIZE * 2);
    unsigned short* vt  = (unsigned short*)alloc(BS * N_HEADS * HEAD_SIZE * 2);     // [B,H,D,S]
    unsigned short* cat = (unsigned short*)alloc(BS * N_HEADS * HEAD_SIZE * 2);     // [B*S, H*D]
    (void)ws_size; (void)in_sizes; (void)n_in; (void)out_size;

    // 1) x -> bf16
    {
        int n = (int)(BS * EMBED);
        cvt_f32_bf16<<<(n + 255) / 256, 256, 0, stream>>>(x, xbf, n);
    }
    // 2) weight transposes -> bf16 (B-operands column-major in memory)
    {
        int n = EMBED * HEAD_SIZE;  // per head
        dim3 g((n + 255) / 256, 1, N_HEADS);
        transpose_cvt<<<g, 256, 0, stream>>>(Wq, wqT, EMBED, HEAD_SIZE);
        transpose_cvt<<<g, 256, 0, stream>>>(Wk, wkT, EMBED, HEAD_SIZE);
        transpose_cvt<<<g, 256, 0, stream>>>(Wv, wvT, EMBED, HEAD_SIZE);
        int m = EMBED * EMBED;
        transpose_cvt<<<dim3((m + 255) / 256, 1, 1), 256, 0, stream>>>(Wo, woT, EMBED, EMBED);
    }
    // 3) QKV projections (z = b*3 + {q,k,v})
    qkv_proj<<<dim3(CTX / 64, N_HEADS, BATCH * 3), 128, 0, stream>>>(
        xbf, wqT, wkT, wvT, qb, kb, vt);
    // 4) causal flash attention
    flash_attn<<<dim3(CTX / 64, N_HEADS, BATCH), 128, 0, stream>>>(qb, kb, vt, cat);
    // 5) output projection (fp32 result)
    out_proj<<<dim3((int)(BS / 64), EMBED / 64, 1), 128, 0, stream>>>(cat, woT, out);
}
